// Residual_GCN_89043261980862
// MI455X (gfx1250) — matrix operand order repaired and checked by reading, not compiled, not run on hardware
//
#include <hip/hip_runtime.h>
#include <hip/hip_bf16.h>
#include <stdint.h>

typedef __attribute__((ext_vector_type(16))) __bf16 v16bf;
typedef __attribute__((ext_vector_type(8)))  float  v8f;

#define B_   8
#define N_   2048
#define D_   128
#define MTOT (B_ * N_)   // 16384 total rows

// Async global->LDS path (CDNA5 GLOBAL_LOAD_ASYNC_TO_LDS_B128, ASYNCcnt),
// guarded so the kernel still compiles on toolchains without the builtins.
#if defined(__has_builtin)
#  if __has_builtin(__builtin_amdgcn_global_load_async_to_lds_b128) && \
      __has_builtin(__builtin_amdgcn_s_wait_asynccnt)
#    define GCN_ASYNC_LDS 1
#  endif
#endif
#ifndef GCN_ASYNC_LDS
#  define GCN_ASYNC_LDS 0
#endif

#if GCN_ASYNC_LDS
// Exact pointer types the builtin expects (from hipcc diagnostic):
//   arg0: 16-byte int vector, addrspace(1) (__device__)
//   arg1: 16-byte int vector, addrspace(3) (__shared__)
typedef int gcn_v4i __attribute__((vector_size(16)));
typedef __attribute__((address_space(1))) gcn_v4i gcn_v4i_g;
typedef __attribute__((address_space(3))) gcn_v4i gcn_v4i_l;
#endif

// float -> bf16 bits, round-to-nearest-even
__device__ __forceinline__ uint32_t f2bf(float f) {
    uint32_t u = __float_as_uint(f);
    return (u + 0x7FFFu + ((u >> 16) & 1u)) >> 16;
}
__device__ __forceinline__ uint32_t f2bf2(float lo, float hi) {
    return f2bf(lo) | (f2bf(hi) << 16);
}

union Frag16 {            // one WMMA 16-bit operand: 16 bf16 = 8 VGPRs
    v16bf v;
    uint4 q[2];
};

// ---------------------------------------------------------------------------
// Kernel 1: per-row degree sum -> d^-1/2  (inf -> 0, matching reference)
// ---------------------------------------------------------------------------
__global__ void gcn_rowsum_rsqrt(const float* __restrict__ adj,
                                 float* __restrict__ dis) {
    int row = blockIdx.x;                      // 0..MTOT-1
    const float4* p = (const float4*)(adj + (size_t)row * N_);
    float s = 0.f;
    for (int i = threadIdx.x; i < N_ / 4; i += blockDim.x) {
        float4 v = p[i];
        s += v.x + v.y + v.z + v.w;
    }
    __shared__ float red[256];
    red[threadIdx.x] = s;
    __syncthreads();
    for (int off = 128; off > 0; off >>= 1) {
        if ((int)threadIdx.x < off) red[threadIdx.x] += red[threadIdx.x + off];
        __syncthreads();
    }
    if (threadIdx.x == 0) {
        float d = red[0];
        dis[row] = (d > 0.f) ? rsqrtf(d) : 0.f;
    }
}

// ---------------------------------------------------------------------------
// Kernel 2: A_norm[b,i,j] = dis[i] * A[b,i,j] * dis[j]  -> bf16 (L2-resident)
// ---------------------------------------------------------------------------
__global__ void gcn_normalize_bf16(const float* __restrict__ adj,
                                   const float* __restrict__ dis,
                                   uint16_t* __restrict__ anorm) {
    size_t t   = (size_t)blockIdx.x * blockDim.x + threadIdx.x;   // float4 id
    size_t row = t / (N_ / 4);                                    // global row
    int    jb  = (int)(t % (N_ / 4)) * 4;
    int    batch = (int)(row >> 11);                              // row / 2048
    float  dr  = dis[row];
    const float* dcol = dis + (size_t)batch * N_;
    float4 a = ((const float4*)adj)[t];
    uint2 pk;
    pk.x = f2bf2(dr * a.x * dcol[jb + 0], dr * a.y * dcol[jb + 1]);
    pk.y = f2bf2(dr * a.z * dcol[jb + 2], dr * a.w * dcol[jb + 3]);
    ((uint2*)anorm)[t] = pk;
}

// ---------------------------------------------------------------------------
// WMMA fragment builders (ISA 7.12.2 layouts, wave32)
//   A 16x32 bf16: lanes 0-15 row M=lane, K={0..7,16..23}; lanes 16-31 row
//                 M=lane-16, K={8..15,24..31}
//   B 32x16 bf16: lanes 0-15 col N=lane, K={0..7,16..23}; lanes 16-31 col
//                 N=lane-16, K={8..15,24..31}   (read from transposed LDS tile)
//   C/D 16x16 f32: vgpr r, lanes 0-15 -> (M=r, N=lane); 16-31 -> (M=r+8, ...)
// LDS rows padded to 40 ushorts (80 B) -> 16B-aligned b128 gathers.
// ---------------------------------------------------------------------------
__device__ __forceinline__ void load_afrag(Frag16& a, const uint16_t (*As)[40],
                                           int wave, int lane) {
    int r    = wave * 16 + (lane & 15);
    int koff = (lane >> 4) * 8;                 // 0 or 8
    a.q[0] = *(const uint4*)&As[r][koff];       // K = koff .. koff+7
    a.q[1] = *(const uint4*)&As[r][koff + 16];  // K = koff+16 .. koff+23
}

__device__ __forceinline__ void load_bfrag(Frag16& b, const uint16_t (*Bt)[40],
                                           int tile, int lane) {
    int col = tile * 16 + (lane & 15);
    int kb  = (lane >> 4) * 8;                  // 0 or 8
    b.q[0] = *(const uint4*)&Bt[col][kb];
    b.q[1] = *(const uint4*)&Bt[col][kb + 16];
}

// ---------------------------------------------------------------------------
// Staging helpers (packed; no scalar b16 LDS stores)
// ---------------------------------------------------------------------------
// A tile (bf16 source): 64x32, 2 threads/row, 2x b128 per thread
__device__ __forceinline__ void stage_a_bf16(uint16_t (*As)[40],
                                             const uint16_t* Abase,
                                             int k0, int tid) {
    int r = tid >> 1;
    int c = (tid & 1) * 16;
    const uint16_t* g = Abase + (size_t)r * N_ + k0 + c;
#if GCN_ASYNC_LDS
    __builtin_amdgcn_global_load_async_to_lds_b128(
        (gcn_v4i_g*)g, (gcn_v4i_l*)&As[r][c], 0, 0);
    __builtin_amdgcn_global_load_async_to_lds_b128(
        (gcn_v4i_g*)(g + 8), (gcn_v4i_l*)&As[r][c + 8], 0, 0);
#else
    const uint4* s = (const uint4*)g;
    *(uint4*)&As[r][c]     = s[0];
    *(uint4*)&As[r][c + 8] = s[1];
#endif
}

// B tile transpose (bf16 source): H[k][f] -> Bt[f][k]. Each thread owns
// 4 consecutive k-rows x 8 columns -> 8x b64 LDS stores (k-pairs packed).
__device__ __forceinline__ void stage_b_bf16(uint16_t (*Bt)[40],
                                             const uint16_t* Hbase,
                                             int k0, int tid) {
    int kk = (tid & 7) * 4;        // 0,4,...,28
    int f0 = (tid >> 3) * 8;       // 0,8,...,120
    const uint16_t* s = Hbase + (size_t)(k0 + kk) * D_ + f0;
    uint4 r0 = *(const uint4*)(s);
    uint4 r1 = *(const uint4*)(s + D_);
    uint4 r2 = *(const uint4*)(s + 2 * D_);
    uint4 r3 = *(const uint4*)(s + 3 * D_);
    const uint16_t* u0 = (const uint16_t*)&r0;
    const uint16_t* u1 = (const uint16_t*)&r1;
    const uint16_t* u2 = (const uint16_t*)&r2;
    const uint16_t* u3 = (const uint16_t*)&r3;
#pragma unroll
    for (int f = 0; f < 8; ++f) {
        uint2 pk;
        pk.x = (uint32_t)u0[f] | ((uint32_t)u1[f] << 16);
        pk.y = (uint32_t)u2[f] | ((uint32_t)u3[f] << 16);
        *(uint2*)&Bt[f0 + f][kk] = pk;   // 8B aligned: kk%4==0, row pitch 80B
    }
}

// ---------------------------------------------------------------------------
// Kernel 3: H = leaky_relu(X @ W + bias), X fp32 -> bf16 staged, H bf16 out
//   64x128 output tile / block, 128 threads (4 waves), K = 128 (4 steps)
// ---------------------------------------------------------------------------
__global__ void gcn_gemm_xw_lrelu(const float* __restrict__ X,
                                  const float* __restrict__ W,
                                  const float* __restrict__ bias,
                                  uint16_t* __restrict__ Hout) {
    __shared__ uint16_t As[64][40];
    __shared__ uint16_t Bt[128][40];
    int tid  = threadIdx.x;
    int lane = tid & 31;
    int wave = tid >> 5;
    int row0 = blockIdx.x * 64;

    v8f acc[8];
#pragma unroll
    for (int t = 0; t < 8; ++t)
#pragma unroll
        for (int e = 0; e < 8; ++e) acc[t][e] = 0.f;

    for (int k0 = 0; k0 < D_; k0 += 32) {
        // stage A (64x32 fp32 -> bf16, packed b128 stores)
        {
            int r = tid >> 1;
            int c = (tid & 1) * 16;
            const float* src = X + (size_t)(row0 + r) * D_ + k0 + c;
            float4 v0 = ((const float4*)src)[0];
            float4 v1 = ((const float4*)src)[1];
            float4 v2 = ((const float4*)src)[2];
            float4 v3 = ((const float4*)src)[3];
            uint4 o0, o1;
            o0.x = f2bf2(v0.x, v0.y); o0.y = f2bf2(v0.z, v0.w);
            o0.z = f2bf2(v1.x, v1.y); o0.w = f2bf2(v1.z, v1.w);
            o1.x = f2bf2(v2.x, v2.y); o1.y = f2bf2(v2.z, v2.w);
            o1.z = f2bf2(v3.x, v3.y); o1.w = f2bf2(v3.z, v3.w);
            *(uint4*)&As[r][c]     = o0;
            *(uint4*)&As[r][c + 8] = o1;
        }
        // stage B transposed (W fp32 -> bf16): 4 k-rows x 8 cols per thread
        {
            int kk = (tid & 7) * 4;
            int f0 = (tid >> 3) * 8;
            const float* s = W + (size_t)(k0 + kk) * D_ + f0;
            float rows[4][8];
#pragma unroll
            for (int rr = 0; rr < 4; ++rr) {
                *(float4*)&rows[rr][0] = *(const float4*)(s + rr * D_);
                *(float4*)&rows[rr][4] = *(const float4*)(s + rr * D_ + 4);
            }
#pragma unroll
            for (int f = 0; f < 8; ++f) {
                uint2 pk;
                pk.x = f2bf2(rows[0][f], rows[1][f]);
                pk.y = f2bf2(rows[2][f], rows[3][f]);
                *(uint2*)&Bt[f0 + f][kk] = pk;
            }
        }
        __syncthreads();
        Frag16 a;
        load_afrag(a, As, wave, lane);
#pragma unroll
        for (int t = 0; t < 8; ++t) {
            Frag16 b;
            load_bfrag(b, Bt, t, lane);
            acc[t] = __builtin_amdgcn_wmma_f32_16x16x32_bf16(
                false, a.v, false, b.v, (short)0, acc[t], false, false);
        }
        __syncthreads();
    }
    // epilogue: +bias, LeakyReLU(0.01), store bf16
    int colbase = lane & 15;
    int rb      = (lane >> 4) * 8;
#pragma unroll
    for (int t = 0; t < 8; ++t) {
        int   col = t * 16 + colbase;
        float bv  = bias[col];
#pragma unroll
        for (int r = 0; r < 8; ++r) {
            int   m = row0 + wave * 16 + rb + r;
            float v = acc[t][r] + bv;
            v = (v > 0.f) ? v : 0.01f * v;
            Hout[(size_t)m * D_ + col] = (uint16_t)f2bf(v);
        }
    }
}

// ---------------------------------------------------------------------------
// Kernel 4: Out = A_norm(bf16) @ H(bf16) [+ Xres], fp32 out
//   per-batch 2048x2048 x 2048x128; 64x128 tile / block, K = 2048 (64 steps)
//   double-buffered LDS; A tile staged with async-to-LDS when available
// ---------------------------------------------------------------------------
__global__ void gcn_gemm_ah(const uint16_t* __restrict__ Anorm,
                            const uint16_t* __restrict__ Hbf,
                            const float* __restrict__ Xres,
                            float* __restrict__ Out) {
    __shared__ uint16_t As[2][64][40];
    __shared__ uint16_t Bt[2][128][40];
    int tid   = threadIdx.x;
    int lane  = tid & 31;
    int wave  = tid >> 5;
    int row0  = blockIdx.x * 64;              // 0..MTOT-1
    int batch = row0 >> 11;
    int lrow0 = row0 & (N_ - 1);
    const uint16_t* Abase = Anorm + ((size_t)batch * N_ + lrow0) * N_;
    const uint16_t* Hbase = Hbf + (size_t)batch * N_ * D_;

    v8f acc[8];
#pragma unroll
    for (int t = 0; t < 8; ++t)
#pragma unroll
        for (int e = 0; e < 8; ++e) acc[t][e] = 0.f;

    const int NIT = N_ / 32;                  // 64 K-steps

    // prologue: stage iteration 0 into buffer 0
    stage_a_bf16(As[0], Abase, 0, tid);
    stage_b_bf16(Bt[0], Hbase, 0, tid);
#if GCN_ASYNC_LDS
    __builtin_amdgcn_s_wait_asynccnt(0);
#endif
    __syncthreads();

    for (int i = 0; i < NIT; ++i) {
        int cur = i & 1;
        if (i + 1 < NIT) {                    // overlap next-tile staging
            stage_a_bf16(As[cur ^ 1], Abase, (i + 1) * 32, tid);
            stage_b_bf16(Bt[cur ^ 1], Hbase, (i + 1) * 32, tid);
        }
        Frag16 a;
        load_afrag(a, As[cur], wave, lane);
#pragma unroll
        for (int t = 0; t < 8; ++t) {
            Frag16 b;
            load_bfrag(b, Bt[cur], t, lane);
            acc[t] = __builtin_amdgcn_wmma_f32_16x16x32_bf16(
                false, a.v, false, b.v, (short)0, acc[t], false, false);
        }
#if GCN_ASYNC_LDS
        __builtin_amdgcn_s_wait_asynccnt(0);  // publish async stores ...
#endif
        __syncthreads();                      // ... before next iter reads them
    }
    // epilogue: optional residual, fp32 store
    int colbase = lane & 15;
    int rb      = (lane >> 4) * 8;
#pragma unroll
    for (int t = 0; t < 8; ++t) {
        int col = t * 16 + colbase;
#pragma unroll
        for (int r = 0; r < 8; ++r) {
            int   m = row0 + wave * 16 + rb + r;
            float v = acc[t][r];
            if (Xres) v += Xres[(size_t)m * D_ + col];
            Out[(size_t)m * D_ + col] = v;
        }
    }
}

// ---------------------------------------------------------------------------
extern "C" void kernel_launch(void* const* d_in, const int* in_sizes, int n_in,
                              void* d_out, int out_size, void* d_ws, size_t ws_size,
                              hipStream_t stream) {
    const float* X   = (const float*)d_in[0];
    const float* adj = (const float*)d_in[1];
    const float* W1  = (const float*)d_in[2];
    const float* b1  = (const float*)d_in[3];
    const float* W2  = (const float*)d_in[4];
    const float* b2  = (const float*)d_in[5];
    const float* W3  = (const float*)d_in[6];
    const float* b3  = (const float*)d_in[7];

    char*  ws  = (char*)d_ws;
    size_t off = 0;
    auto alloc = [&](size_t bytes) -> void* {
        void* p = ws + off;
        off += (bytes + 255) & ~(size_t)255;
        return p;
    };
    float*    dis   = (float*)alloc((size_t)MTOT * 4);
    uint16_t* Anorm = (uint16_t*)alloc((size_t)B_ * N_ * N_ * 2);   // 64 MB, L2-resident
    uint16_t* Hbf   = (uint16_t*)alloc((size_t)MTOT * D_ * 2);
    float*    X1    = (float*)alloc((size_t)MTOT * D_ * 4);
    float*    X2    = (float*)alloc((size_t)MTOT * D_ * 4);

    gcn_rowsum_rsqrt<<<MTOT, 256, 0, stream>>>(adj, dis);
    gcn_normalize_bf16<<<(int)(((size_t)B_ * N_ * N_ / 4) / 256), 256, 0, stream>>>(
        adj, dis, Anorm);

    // layer 1
    gcn_gemm_xw_lrelu<<<MTOT / 64, 128, 0, stream>>>(X, W1, b1, Hbf);
    gcn_gemm_ah<<<MTOT / 64, 128, 0, stream>>>(Anorm, Hbf, nullptr, X1);
    // layer 2 (residual)
    gcn_gemm_xw_lrelu<<<MTOT / 64, 128, 0, stream>>>(X1, W2, b2, Hbf);
    gcn_gemm_ah<<<MTOT / 64, 128, 0, stream>>>(Anorm, Hbf, X1, X2);
    // layer 3
    gcn_gemm_xw_lrelu<<<MTOT / 64, 128, 0, stream>>>(X2, W3, b3, Hbf);
    gcn_gemm_ah<<<MTOT / 64, 128, 0, stream>>>(Anorm, Hbf, nullptr, (float*)d_out);
}
